// RecDQN_48258252538293
// MI455X (gfx1250) — compile-verified, matrix operand order checked by reading, not run on hardware
//
#include <hip/hip_runtime.h>
#include <hip/hip_bf16.h>
#include <math.h>

typedef _Float16 f16;
typedef __attribute__((ext_vector_type(16))) _Float16 v16h;
typedef __attribute__((ext_vector_type(8)))  _Float16 v8h;
typedef __attribute__((ext_vector_type(8)))  float    v8f;

#define B_TOTAL 32768
#define MTILE   16
#define TSTEPS  32

// f16 workspace layout (element offsets)
#define OFF_WHH 0          // [512][160]  W_hh | W_ih | (b_ih+b_hh)@134 | 0
#define OFF_WTS 81920      // [128][160]  W_ts | b_ts@134 | 0
#define OFF_WC1 102400     // [128][288]  W_c1 | b_c1@256 | 0
#define OFF_WC2 139264     // [16][160]   W_c2 (rows 0..8) | b_c2@134 | 0
#define OFF_WOS 141824     // [128][32]   W_os | b_os@8 | 0
#define WS_ELTS 145920

#define HA_STRIDE 168      // 16 rows x 160 K (+8 pad)  : LSTM A / c2 A
#define CA_STRIDE 296      // 16 rows x 288 K (+8 pad)  : c1 A
#define XT_STRIDE 192      // 16 rows x 32*6 cleaned TS obs
#define OS_STRIDE 40       // 16 rows x 32 K (+8 pad)   : os A

#define WMMA(a, b, c) __builtin_amdgcn_wmma_f32_16x16x32_f16( \
    false, (a), false, (b), (short)0, (c), false, false)

__device__ __forceinline__ v16h ld_ab(const f16* p) {
  // per-lane A/B tile fragment: 8 f16 at k0, 8 f16 at k0+16
  v8h lo = *(const v8h*)p;
  v8h hi = *(const v8h*)(p + 16);
  return __builtin_shufflevector(lo, hi, 0,1,2,3,4,5,6,7,8,9,10,11,12,13,14,15);
}

__device__ __forceinline__ float fast_rcp(float x) {
  return __builtin_amdgcn_rcpf(x);
}
__device__ __forceinline__ float fast_sigmoid(float x) {
  // 1/(1+exp(-x)) : v_mul + v_exp_f32 + v_add + v_rcp_f32
  return fast_rcp(1.0f + __expf(-x));
}
__device__ __forceinline__ float fast_tanh(float x) {
#if __has_builtin(__builtin_amdgcn_tanhf)
  return __builtin_amdgcn_tanhf(x);      // native V_TANH_F32 on gfx1250
#else
  float e = __expf(2.0f * x);            // exp-based fallback, branch-free
  return 1.0f - 2.0f * fast_rcp(e + 1.0f);
#endif
}

// ---------------- weight prep: fp32 -> augmented f16 matrices ----------------
__global__ void prep_kernel(const float* __restrict__ Wos, const float* __restrict__ bos,
                            const float* __restrict__ Wih, const float* __restrict__ Whh,
                            const float* __restrict__ bih, const float* __restrict__ bhh,
                            const float* __restrict__ Wts, const float* __restrict__ bts,
                            const float* __restrict__ Wc1, const float* __restrict__ bc1,
                            const float* __restrict__ Wc2, const float* __restrict__ bc2,
                            f16* __restrict__ ws) {
  int i = blockIdx.x * blockDim.x + threadIdx.x;
  int stride = gridDim.x * blockDim.x;
  for (; i < WS_ELTS; i += stride) {
    float v = 0.0f;
    if (i < OFF_WTS) {                       // W_hh augmented
      int g = i / 160, k = i % 160;
      if (k < 128)       v = Whh[g*128 + k];
      else if (k < 134)  v = Wih[g*6 + (k - 128)];
      else if (k == 134) v = bih[g] + bhh[g];
    } else if (i < OFF_WC1) {                // W_ts augmented
      int j = i - OFF_WTS; int o = j / 160, k = j % 160;
      if (k < 128)       v = Wts[o*128 + k];
      else if (k == 134) v = bts[o];
    } else if (i < OFF_WC2) {                // W_c1 augmented
      int j = i - OFF_WC1; int o = j / 288, k = j % 288;
      if (k < 256)       v = Wc1[o*256 + k];
      else if (k == 256) v = bc1[o];
    } else if (i < OFF_WOS) {                // W_c2 augmented (padded to 16 rows)
      int j = i - OFF_WC2; int o = j / 160, k = j % 160;
      if (o < 9) {
        if (k < 128)       v = Wc2[o*128 + k];
        else if (k == 134) v = bc2[o];
      }
    } else {                                 // W_os augmented
      int j = i - OFF_WOS; int o = j / 32, k = j % 32;
      if (k < 8)        v = Wos[o*8 + k];
      else if (k == 8)  v = bos[o];
    }
    ws[i] = (f16)v;
  }
}

// ---------------- fused RecDQN forward: one workgroup = 16 batch rows --------
__global__ __launch_bounds__(256) void recdqn_kernel(
    const float* __restrict__ s, const f16* __restrict__ ws,
    float* __restrict__ out) {
  __shared__ __align__(16) f16 hA[MTILE * HA_STRIDE];   // [h | x_t | 1 | 0] A-matrix
  __shared__ __align__(16) f16 cA[MTILE * CA_STRIDE];   // [x_OS | x_TS | 1 | 0]
  __shared__ __align__(16) f16 xts[MTILE * XT_STRIDE];  // cleaned TS obs
  __shared__ __align__(16) f16 osA[MTILE * OS_STRIDE];  // [s_OS | 1 | 0]
  __shared__ int hidx[MTILE];

  const int tid   = threadIdx.x;
  const int wv    = tid >> 5;          // wave 0..7 -> owns hidden cols [16w,16w+16)
  const int lane  = tid & 31;
  const int lhalf = (lane >> 4);       // 0 or 1
  const int lmod  = lane & 15;
  const int b0    = blockIdx.x * MTILE;

  // ---- LDS init ----
  for (int i = tid; i < MTILE*HA_STRIDE; i += 256) hA[i] = (f16)0.0f;
  for (int i = tid; i < MTILE*CA_STRIDE; i += 256) cA[i] = (f16)0.0f;
  for (int i = tid; i < MTILE*XT_STRIDE; i += 256) {
    int m = i / XT_STRIDE, j = i % XT_STRIDE;
    float v = s[(b0 + m)*200 + 8 + j];
    xts[m*XT_STRIDE + j] = (f16)((v != v) ? 0.0f : v);
  }
  for (int i = tid; i < MTILE*OS_STRIDE; i += 256) {
    int m = i / OS_STRIDE, k = i % OS_STRIDE;
    float v = (k < 8) ? s[(b0 + m)*200 + k] : (k == 8 ? 1.0f : 0.0f);
    osA[m*OS_STRIDE + k] = (f16)v;
  }
  if (tid < MTILE) {
    hA[tid*HA_STRIDE + 134] = (f16)1.0f;   // bias column for LSTM / ts / c2
    cA[tid*CA_STRIDE + 256] = (f16)1.0f;   // bias column for c1
    int n = 0;
    for (int t = 0; t < TSTEPS; ++t) {
      float v = s[(b0 + tid)*200 + 8 + t*6];
      n += (v == v) ? 1 : 0;
    }
    hidx[tid] = (n > 0) ? (n - 1) : -1;    // -1: never select -> selh stays 0
  }
  __syncthreads();

  // ---- preload this wave's W_hh_aug B-tiles into registers (20 tiles) ----
  v16h Bh[4][5];
  {
    const f16* whh = ws + OFF_WHH;
#pragma unroll
    for (int gb = 0; gb < 4; ++gb) {
#pragma unroll
      for (int kc = 0; kc < 5; ++kc) {
        const f16* p = whh + (gb*128 + wv*16 + lmod)*160 + kc*32 + lhalf*8;
        Bh[gb][kc] = ld_ab(p);
      }
    }
  }
  int hsel[8];
#pragma unroll
  for (int j = 0; j < 8; ++j) hsel[j] = hidx[j + 8*lhalf];

  v8f cstate = {};
  v8f selh   = {};

  // ---- LSTM recurrence ----
  for (int t = 0; t < TSTEPS; ++t) {
    if (tid < 96) {                         // stage x_t into A columns 128..133
      int m = tid / 6, i = tid % 6;
      hA[m*HA_STRIDE + 128 + i] = xts[m*XT_STRIDE + t*6 + i];
    }
    __syncthreads();

    v8f acc0 = {}, acc1 = {}, acc2 = {}, acc3 = {};
#pragma unroll
    for (int kc = 0; kc < 5; ++kc) {
      v16h a = ld_ab(&hA[lmod*HA_STRIDE + kc*32 + lhalf*8]);
      acc0 = WMMA(a, Bh[0][kc], acc0);      // i gates
      acc1 = WMMA(a, Bh[1][kc], acc1);      // f gates
      acc2 = WMMA(a, Bh[2][kc], acc2);      // g gates
      acc3 = WMMA(a, Bh[3][kc], acc3);      // o gates
    }
    __syncthreads();                        // all A reads done before h rewrite

#pragma unroll
    for (int j = 0; j < 8; ++j) {
      float ig = fast_sigmoid(acc0[j]);
      float fg = fast_sigmoid(acc1[j]);
      float gg = fast_tanh(acc2[j]);
      float og = fast_sigmoid(acc3[j]);
      float c  = fg * cstate[j] + ig * gg;
      cstate[j] = c;
      float h  = og * fast_tanh(c);
      selh[j]  = (t == hsel[j]) ? h : selh[j];
      int m = j + 8*lhalf;
      hA[m*HA_STRIDE + wv*16 + lmod] = (f16)h;
    }
  }

  // ---- stage selected h for the ts layer (reuse hA) ----
#pragma unroll
  for (int j = 0; j < 8; ++j) {
    int m = j + 8*lhalf;
    hA[m*HA_STRIDE + wv*16 + lmod] = (f16)selh[j];
  }
  if (tid < 96) {                           // clear x_t columns; bias col stays 1
    int m = tid / 6, i = tid % 6;
    hA[m*HA_STRIDE + 128 + i] = (f16)0.0f;
  }
  __syncthreads();

  // ---- x_TS = relu(selh @ W_ts^T + b_ts) -> cA cols 128..255 ----
  {
    const f16* wts = ws + OFF_WTS;
    v8f acc = {};
#pragma unroll
    for (int kc = 0; kc < 5; ++kc) {
      v16h a = ld_ab(&hA[lmod*HA_STRIDE + kc*32 + lhalf*8]);
      v16h b = ld_ab(wts + (wv*16 + lmod)*160 + kc*32 + lhalf*8);
      acc = WMMA(a, b, acc);
    }
#pragma unroll
    for (int j = 0; j < 8; ++j) {
      int m = j + 8*lhalf;
      float v = acc[j] > 0.0f ? acc[j] : 0.0f;
      cA[m*CA_STRIDE + 128 + wv*16 + lmod] = (f16)v;
    }
  }

  // ---- x_OS = relu(s_OS @ W_os^T + b_os) -> cA cols 0..127 ----
  {
    const f16* wos = ws + OFF_WOS;
    v16h a = ld_ab(&osA[lmod*OS_STRIDE + lhalf*8]);
    v16h b = ld_ab(wos + (wv*16 + lmod)*32 + lhalf*8);
    v8f acc = {};
    acc = WMMA(a, b, acc);
#pragma unroll
    for (int j = 0; j < 8; ++j) {
      int m = j + 8*lhalf;
      float v = acc[j] > 0.0f ? acc[j] : 0.0f;
      cA[m*CA_STRIDE + wv*16 + lmod] = (f16)v;
    }
  }
  __syncthreads();

  // ---- c1 = relu(cA @ W_c1^T + b_c1) -> hA cols 0..127 ----
  {
    const f16* wc1 = ws + OFF_WC1;
    v8f acc = {};
#pragma unroll
    for (int kc = 0; kc < 9; ++kc) {
      v16h a = ld_ab(&cA[lmod*CA_STRIDE + kc*32 + lhalf*8]);
      v16h b = ld_ab(wc1 + (wv*16 + lmod)*288 + kc*32 + lhalf*8);
      acc = WMMA(a, b, acc);
    }
#pragma unroll
    for (int j = 0; j < 8; ++j) {
      int m = j + 8*lhalf;
      float v = acc[j] > 0.0f ? acc[j] : 0.0f;
      hA[m*HA_STRIDE + wv*16 + lmod] = (f16)v;
    }
  }
  __syncthreads();

  // ---- q = c1 @ W_c2^T + b_c2  (wave 0 only; N padded 9->16) ----
  if (wv == 0) {
    const f16* wc2 = ws + OFF_WC2;
    v8f acc = {};
#pragma unroll
    for (int kc = 0; kc < 5; ++kc) {
      v16h a = ld_ab(&hA[lmod*HA_STRIDE + kc*32 + lhalf*8]);
      v16h b = ld_ab(wc2 + lmod*160 + kc*32 + lhalf*8);
      acc = WMMA(a, b, acc);
    }
#pragma unroll
    for (int j = 0; j < 8; ++j) {
      int m = j + 8*lhalf;
      if (lmod < 9) out[(b0 + m)*9 + lmod] = acc[j];
    }
  }
}

extern "C" void kernel_launch(void* const* d_in, const int* in_sizes, int n_in,
                              void* d_out, int out_size, void* d_ws, size_t ws_size,
                              hipStream_t stream) {
  const float* s   = (const float*)d_in[0];
  const float* Wos = (const float*)d_in[1];
  const float* bos = (const float*)d_in[2];
  const float* Wih = (const float*)d_in[3];
  const float* Whh = (const float*)d_in[4];
  const float* bih = (const float*)d_in[5];
  const float* bhh = (const float*)d_in[6];
  const float* Wts = (const float*)d_in[7];
  const float* bts = (const float*)d_in[8];
  const float* Wc1 = (const float*)d_in[9];
  const float* bc1 = (const float*)d_in[10];
  const float* Wc2 = (const float*)d_in[11];
  const float* bc2 = (const float*)d_in[12];
  f16* ws    = (f16*)d_ws;
  float* out = (float*)d_out;

  prep_kernel<<<(WS_ELTS + 255)/256, 256, 0, stream>>>(
      Wos, bos, Wih, Whh, bih, bhh, Wts, bts, Wc1, bc1, Wc2, bc2, ws);
  recdqn_kernel<<<B_TOTAL/MTILE, 256, 0, stream>>>(s, ws, out);
}
